// MultiContextPerception_12146167513383
// MI455X (gfx1250) — matrix operand order repaired and checked by reading, not compiled, not run on hardware
//
#include <hip/hip_runtime.h>

typedef __bf16 bf16;
typedef __attribute__((ext_vector_type(16))) __bf16 v16bf;
typedef __attribute__((ext_vector_type(8)))  __bf16 v8bf;
typedef __attribute__((ext_vector_type(8)))  float  v8f;

#define LLEN 1024
#define BATCH 8
#define DMODEL 512
#define NH 8
#define HDIM 64
#define LTXT 32
#define MTOK (LLEN*BATCH)   // 8192

static __device__ inline v8bf ld8(const bf16* p) { return *(const v8bf*)p; }
static __device__ inline v16bf cat8(v8bf a, v8bf b) {
    return __builtin_shufflevector(a, b, 0,1,2,3,4,5,6,7,8,9,10,11,12,13,14,15);
}

// ---------------------------------------------------------------- convert f32 -> bf16
__global__ void k_cvt(const float* __restrict__ in, bf16* __restrict__ out, long n) {
    long i = (long)blockIdx.x * blockDim.x + threadIdx.x;
    if (i < n) out[i] = (bf16)in[i];
}

// ---------------------------------------------------------------- sentence pooling
__global__ void __launch_bounds__(256)
k_sent(const float* __restrict__ txt, const float* __restrict__ wtsw,
       const float* __restrict__ wtsb, const unsigned char* __restrict__ tmask,
       float* __restrict__ sent) {
    __shared__ float aw[LTXT];
    int b = blockIdx.x, t = threadIdx.x;
    if (t < LTXT) {
        float dot = wtsb[0];
        const float* tr = txt + ((long)t * BATCH + b) * DMODEL;
        for (int d = 0; d < DMODEL; ++d) dot += tr[d] * wtsw[d];
        aw[t] = tmask[b * LTXT + t] ? dot : -1e9f;
    }
    __syncthreads();
    if (t == 0) {
        float mx = -1e30f;
        for (int i = 0; i < LTXT; ++i) mx = fmaxf(mx, aw[i]);
        float s = 0.f;
        for (int i = 0; i < LTXT; ++i) { aw[i] = __expf(aw[i] - mx); s += aw[i]; }
        float inv = 1.f / s;
        for (int i = 0; i < LTXT; ++i) aw[i] *= inv;
    }
    __syncthreads();
    for (int d = t; d < DMODEL; d += 256) {
        float acc = 0.f;
        for (int lt = 0; lt < LTXT; ++lt)
            acc += aw[lt] * txt[((long)lt * BATCH + b) * DMODEL + d];
        sent[b * DMODEL + d] = acc;
    }
}

// ---------------------------------------------------------------- relevance top-k -> ec mask (1=ignore)
__global__ void __launch_bounds__(256)
k_topk(const float* __restrict__ x, const float* __restrict__ sent,
       const unsigned char* __restrict__ vmask, float* __restrict__ maskec) {
    __shared__ float sc[LLEN];
    __shared__ int   si[LLEN];
    int b = blockIdx.x, t = threadIdx.x;
    for (int i = t; i < LLEN; i += 256) {
        float dot = 0.f;
        const float* xr = x + ((long)i * BATCH + b) * DMODEL;
        const float* sr = sent + b * DMODEL;
        for (int d = 0; d < DMODEL; ++d) dot += xr[d] * sr[d];
        sc[i] = vmask[b * LLEN + i] ? dot : -1e9f;
        si[i] = i;
    }
    __syncthreads();
    for (int k = 2; k <= LLEN; k <<= 1)
        for (int j = k >> 1; j > 0; j >>= 1) {
            for (int i = t; i < LLEN; i += 256) {
                int ix = i ^ j;
                if (ix > i) {
                    bool dec = ((i & k) == 0);
                    float va = sc[i], vb = sc[ix];
                    int ia = si[i], ib = si[ix];
                    bool aLess = (va < vb) || (va == vb && ia > ib);
                    bool sw = dec ? aLess : !aLess;
                    if (sw) { sc[i] = vb; sc[ix] = va; si[i] = ib; si[ix] = ia; }
                }
            }
            __syncthreads();
        }
    for (int i = t; i < LLEN; i += 256) maskec[b * LLEN + i] = 1.f;
    __syncthreads();
    for (int r = t; r < LLEN / 2; r += 256) {
        int l = si[r];
        if (vmask[b * LLEN + l]) maskec[b * LLEN + l] = 0.f;
    }
}

// ---------------------------------------------------------------- multi-stride max pool
__global__ void k_pool(const float* __restrict__ x, const unsigned char* __restrict__ vmask,
                       bf16* __restrict__ keys, float* __restrict__ maskIg,
                       int s1, int s2, int Lp1, int Lk) {
    long i = (long)blockIdx.x * blockDim.x + threadIdx.x;
    long total = (long)Lk * BATCH * DMODEL;
    if (i >= total) return;
    int d = (int)(i % DMODEL);
    int b = (int)((i / DMODEL) % BATCH);
    int lp = (int)(i / (DMODEL * BATCH));
    int s, base;
    if (lp < Lp1) { s = s1; base = lp * s1; } else { s = s2; base = (lp - Lp1) * s2; }
    float m = -1e30f;
    for (int j = 0; j < s; ++j) {
        int pos = base + j;
        float v = (pos < LLEN) ? x[((long)pos * BATCH + b) * DMODEL + d] : 0.f;  // zero pad joins max
        m = fmaxf(m, v);
    }
    keys[i] = (bf16)m;
    if (d == 0) {
        float mm = 0.f;
        for (int j = 0; j < s; ++j) {
            int pos = base + j;
            float v = (pos < LLEN) ? (vmask[b * LLEN + pos] ? 1.f : 0.f) : 0.f;
            mm = fmaxf(mm, v);
        }
        maskIg[b * Lk + lp] = (mm > 0.f) ? 0.f : 1.f;
    }
}

// ---------------------------------------------------------------- tiled bf16 WMMA GEMM: C = A(M,K) * W(N,K)^T
// explicit ping-pong double buffer, K must be a multiple of 64 (512/2048 here):
// no register rotation copies across the loop back-edge.
__global__ void __launch_bounds__(256)
k_gemm(const bf16* __restrict__ A, const bf16* __restrict__ W,
       const float* __restrict__ bias, const float* __restrict__ addC,
       float* __restrict__ outF, bf16* __restrict__ outB,
       int M, int N, int K, int relu) {
    int lane = threadIdx.x & 31;
    int w = threadIdx.x >> 5;
    int wm = w & 3, wn = w >> 2;          // 4 waves in M, 2 in N
    int m0 = blockIdx.y * 64 + wm * 16;
    int n0 = blockIdx.x * 128 + wn * 64;
    int hf = lane >> 4, l16 = lane & 15;

    v8f acc[4];
    for (int nt = 0; nt < 4; ++nt)
        for (int r = 0; r < 8; ++r) acc[nt][r] = 0.f;

    int rowA = m0 + l16;
    // clamp OOB rows/cols to valid memory: their products only pollute
    // C rows/cols that the bounds-checked epilogue never stores.
    const bf16* arow = A + (long)(rowA < M ? rowA : 0) * K;
    const bf16* wrow[4];
#pragma unroll
    for (int nt = 0; nt < 4; ++nt) {
        int col = n0 + nt * 16 + l16;
        wrow[nt] = W + (long)(col < N ? col : 0) * K;
    }

    // A fragment: one contiguous 16-element run per lane (k = 16*hf + s)
    // B fragment: one contiguous 16-element run per lane (k = 16*hf + s, same order as A)
    auto loadA = [&](int k0) -> v16bf {
        return cat8(ld8(arow + k0 + hf * 16), ld8(arow + k0 + hf * 16 + 8));
    };
    auto loadBs = [&](int k0, v16bf* bb) {
#pragma unroll
        for (int nt = 0; nt < 4; ++nt) {
            const bf16* wp = wrow[nt] + k0 + hf * 16;
            bb[nt] = cat8(ld8(wp), ld8(wp + 8));
        }
    };

    v16bf aP, bP[4], aQ, bQ[4];
    aP = loadA(0);
    loadBs(0, bP);
    for (int k0 = 0; k0 < K; k0 += 64) {
        aQ = loadA(k0 + 32);
        loadBs(k0 + 32, bQ);
        if (k0 + 128 <= K) __builtin_prefetch(arow + k0 + 128, 0, 1);
#pragma unroll
        for (int nt = 0; nt < 4; ++nt)
            acc[nt] = __builtin_amdgcn_wmma_f32_16x16x32_bf16(
                false, aP, false, bP[nt], (short)0, acc[nt], false, false);
        if (k0 + 64 < K) {
            aP = loadA(k0 + 64);
            loadBs(k0 + 64, bP);
        }
#pragma unroll
        for (int nt = 0; nt < 4; ++nt)
            acc[nt] = __builtin_amdgcn_wmma_f32_16x16x32_bf16(
                false, aQ, false, bQ[nt], (short)0, acc[nt], false, false);
    }
    for (int nt = 0; nt < 4; ++nt) {
        int col = n0 + nt * 16 + l16;
        if (col >= N) continue;
        float bv = bias ? bias[col] : 0.f;
        for (int r = 0; r < 8; ++r) {
            int row = m0 + r + hf * 8;
            if (row >= M) continue;
            long idx = (long)row * N + col;
            float v = acc[nt][r] + bv;
            if (addC) v += addC[idx];
            if (relu) v = fmaxf(v, 0.f);
            if (outF) outF[idx] = v;
            if (outB) outB[idx] = (bf16)v;
        }
    }
}

// ---------------------------------------------------------------- pack Q: (l,b,*) -> (b*H+h, l, hd)
__global__ void k_packq(const bf16* __restrict__ src, int srcN, int colOff,
                        bf16* __restrict__ dst, int Llen) {
    long i = (long)blockIdx.x * blockDim.x + threadIdx.x;
    long total = (long)BATCH * NH * Llen * HDIM;
    if (i >= total) return;
    int d = (int)(i & 63);
    long r = i >> 6;
    int l = (int)(r % Llen);
    long bh = r / Llen;
    int b = (int)(bh / NH), h = (int)(bh % NH);
    dst[i] = src[((long)l * BATCH + b) * srcN + colOff + h * HDIM + d];
}

// ---------------------------------------------------------------- pack K: (l,b,*) -> (bh, key, hd), zero pad to Lkpad
__global__ void k_packk(const bf16* __restrict__ src, int srcN, int colOff,
                        bf16* __restrict__ dst, int Lk, int Lkpad) {
    long i = (long)blockIdx.x * blockDim.x + threadIdx.x;
    long total = (long)BATCH * NH * Lkpad * HDIM;
    if (i >= total) return;
    int d = (int)(i & 63);
    long r = i >> 6;
    int l = (int)(r % Lkpad);
    long bh = r / Lkpad;
    int b = (int)(bh / NH), h = (int)(bh % NH);
    dst[i] = (l < Lk) ? src[((long)l * BATCH + b) * srcN + colOff + h * HDIM + d] : (bf16)0.f;
}

// ---------------------------------------------------------------- pack V transposed: (l,b,*) -> (bh, hd, key), zero pad
__global__ void k_packvt(const bf16* __restrict__ src, int srcN, int colOff,
                         bf16* __restrict__ dst, int Lk, int Lkpad) {
    long i = (long)blockIdx.x * blockDim.x + threadIdx.x;
    long total = (long)BATCH * NH * HDIM * Lkpad;
    if (i >= total) return;
    int key = (int)(i % Lkpad);
    long r = i / Lkpad;
    int d = (int)(r & 63);
    long bh = r >> 6;
    int b = (int)(bh / NH), h = (int)(bh % NH);
    dst[i] = (key < Lk) ? src[((long)key * BATCH + b) * srcN + colOff + h * HDIM + d] : (bf16)0.f;
}

// ---------------------------------------------------------------- flash attention: one wave per 16-query tile
// explicit ping-pong K-fragment banks over 64-key macro-steps (Lkpad % 64 == 0);
// V loads issued first inside each tile so softmax VALU hides their latency.
__global__ void __launch_bounds__(32)
k_attn(const bf16* __restrict__ Q, const bf16* __restrict__ Km, const bf16* __restrict__ Vt,
       const float* __restrict__ maskIg, float* __restrict__ O,
       int Lq, int Lk, int Lkpad, float scale) {
    __shared__ float pshare[16 * 32];
    int lane = threadIdx.x;
    int bh = blockIdx.y;
    int b = bh / NH;
    int q0 = blockIdx.x * 16;
    int hf = lane >> 4, l16 = lane & 15;

    const bf16* qbase = Q + ((long)bh * Lq + q0 + l16) * HDIM;
    v16bf a0 = cat8(ld8(qbase + hf * 16),      ld8(qbase + hf * 16 + 8));
    v16bf a1 = cat8(ld8(qbase + 32 + hf * 16), ld8(qbase + 32 + hf * 16 + 8));

    float mrow[8], lrow[8];
    v8f o0, o1, o2, o3;
    for (int r = 0; r < 8; ++r) {
        mrow[r] = -1e30f; lrow[r] = 0.f;
        o0[r] = 0.f; o1[r] = 0.f; o2[r] = 0.f; o3[r] = 0.f;
    }

    // K fragment loader: kf[t*2+c] = d-chunk c of key tile t (d = 32*c + 16*hf + s, same order as Q)
    auto loadK = [&](int kt, v16bf* kf) {
#pragma unroll
        for (int t = 0; t < 2; ++t) {
            const bf16* kb = Km + ((long)bh * Lkpad + kt + t * 16 + l16) * HDIM;
            kf[t * 2 + 0] = cat8(ld8(kb + hf * 16),      ld8(kb + hf * 16 + 8));
            kf[t * 2 + 1] = cat8(ld8(kb + 32 + hf * 16), ld8(kb + 32 + hf * 16 + 8));
        }
    };

    // one 32-key tile: V loads, S = QK^T, mask+online softmax, P stage, O += P V
    auto processTile = [&](int kt, v16bf* kf) {
        v16bf vf[4];
#pragma unroll
        for (int nt = 0; nt < 4; ++nt) {
            const bf16* vp = Vt + ((long)bh * HDIM + nt * 16 + l16) * Lkpad + kt + hf * 16;
            vf[nt] = cat8(ld8(vp), ld8(vp + 8));
        }
        v8f s[2];
#pragma unroll
        for (int t = 0; t < 2; ++t) {
            int key = kt + t * 16 + l16;
            v8f accs;
            for (int r = 0; r < 8; ++r) accs[r] = 0.f;
            accs = __builtin_amdgcn_wmma_f32_16x16x32_bf16(false, a0, false, kf[t * 2 + 0], (short)0, accs, false, false);
            accs = __builtin_amdgcn_wmma_f32_16x16x32_bf16(false, a1, false, kf[t * 2 + 1], (short)0, accs, false, false);
            float mg = (key < Lk) ? maskIg[(long)b * Lk + key] : 1.0f;
            for (int r = 0; r < 8; ++r)
                accs[r] = (mg > 0.5f) ? -1e30f : accs[r] * scale;
            s[t] = accs;
        }
        // online softmax (C layout: lanes<16 -> M=r, lanes>=16 -> M=r+8)
        float nm[8];
        for (int r = 0; r < 8; ++r) {
            float t = fmaxf(s[0][r], s[1][r]);
            for (int m = 1; m <= 8; m <<= 1) t = fmaxf(t, __shfl_xor(t, m, 32));
            nm[r] = fmaxf(mrow[r], t);
        }
        float p0[8], p1[8];
        for (int r = 0; r < 8; ++r) {
            float base = (nm[r] <= -1e29f) ? 0.f : nm[r];
            p0[r] = (s[0][r] <= -1e29f) ? 0.f : __expf(s[0][r] - base);
            p1[r] = (s[1][r] <= -1e29f) ? 0.f : __expf(s[1][r] - base);
            float t = p0[r] + p1[r];
            for (int m = 1; m <= 8; m <<= 1) t += __shfl_xor(t, m, 32);
            float corr = (mrow[r] <= -1e29f) ? 0.f : __expf(mrow[r] - nm[r]);
            lrow[r] = lrow[r] * corr + t;
            mrow[r] = nm[r];
            o0[r] *= corr; o1[r] *= corr; o2[r] *= corr; o3[r] *= corr;
        }
        // stage P (C layout) to LDS, reread in A layout
        __syncthreads();
        for (int r = 0; r < 8; ++r) {
            int row = r + hf * 8;
            pshare[row * 32 + l16]      = p0[r];
            pshare[row * 32 + 16 + l16] = p1[r];
        }
        __syncthreads();
        v16bf pa;
#pragma unroll
        for (int j = 0; j < 16; ++j) {
            int kd = j + hf * 16;
            pa[j] = (bf16)pshare[l16 * 32 + kd];
        }
        o0 = __builtin_amdgcn_wmma_f32_16x16x32_bf16(false, pa, false, vf[0], (short)0, o0, false, false);
        o1 = __builtin_amdgcn_wmma_f32_16x16x32_bf16(false, pa, false, vf[1], (short)0, o1, false, false);
        o2 = __builtin_amdgcn_wmma_f32_16x16x32_bf16(false, pa, false, vf[2], (short)0, o2, false, false);
        o3 = __builtin_amdgcn_wmma_f32_16x16x32_bf16(false, pa, false, vf[3], (short)0, o3, false, false);
    };

    v16bf kA[4], kB[4];
    loadK(0, kA);
    for (int kt = 0; kt < Lkpad; kt += 64) {      // Lkpad % 64 == 0
        loadK(kt + 32, kB);
        if (kt + 64 < Lkpad)
            __builtin_prefetch(Km + ((long)bh * Lkpad + kt + 64 + l16) * HDIM, 0, 1);
        processTile(kt, kA);
        if (kt + 64 < Lkpad) loadK(kt + 64, kA);
        processTile(kt + 32, kB);
    }

    for (int r = 0; r < 8; ++r) {
        float inv = (lrow[r] > 0.f) ? 1.f / lrow[r] : 0.f;
        int row = q0 + r + hf * 8;
        long base = ((long)bh * Lq + row) * HDIM;
        O[base + 0 * 16 + l16] = o0[r] * inv;
        O[base + 1 * 16 + l16] = o1[r] * inv;
        O[base + 2 * 16 + l16] = o2[r] * inv;
        O[base + 3 * 16 + l16] = o3[r] * inv;
    }
}

// ---------------------------------------------------------------- (bh,l,hd) f32 -> (l*B+b, D) bf16
__global__ void k_packo(const float* __restrict__ o, bf16* __restrict__ dst, int Lq) {
    long i = (long)blockIdx.x * blockDim.x + threadIdx.x;
    long total = (long)Lq * BATCH * DMODEL;
    if (i >= total) return;
    int c = (int)(i % DMODEL);
    long r = i / DMODEL;
    int b = (int)(r % BATCH);
    int l = (int)(r / BATCH);
    int h = c >> 6, d = c & 63;
    dst[i] = (bf16)o[(((long)b * NH + h) * Lq + l) * HDIM + d];
}

// ---------------------------------------------------------------- elementwise add (f32 + bf16 out)
__global__ void k_add(const float* __restrict__ a, const float* __restrict__ b,
                      float* __restrict__ outF, bf16* __restrict__ outB, long n) {
    long i = (long)blockIdx.x * blockDim.x + threadIdx.x;
    if (i < n) { float v = a[i] + b[i]; outF[i] = v; outB[i] = (bf16)v; }
}

// ---------------------------------------------------------------- LayerNorm over D=512: LN(a + b)
__global__ void __launch_bounds__(256)
k_ln(const float* __restrict__ a, const float* __restrict__ b,
     const float* __restrict__ g, const float* __restrict__ beta,
     float* __restrict__ outF, bf16* __restrict__ outB) {
    __shared__ float red[256];
    long row = blockIdx.x;
    int t = threadIdx.x;
    long base = row * DMODEL;
    float v0 = a[base + t]       + (b ? b[base + t]       : 0.f);
    float v1 = a[base + 256 + t] + (b ? b[base + 256 + t] : 0.f);
    red[t] = v0 + v1; __syncthreads();
    for (int st = 128; st > 0; st >>= 1) { if (t < st) red[t] += red[t + st]; __syncthreads(); }
    float mu = red[0] / (float)DMODEL; __syncthreads();
    float d0 = v0 - mu, d1 = v1 - mu;
    red[t] = d0 * d0 + d1 * d1; __syncthreads();
    for (int st = 128; st > 0; st >>= 1) { if (t < st) red[t] += red[t + st]; __syncthreads(); }
    float rstd = rsqrtf(red[0] / (float)DMODEL + 1e-5f);
    float o0 = d0 * rstd * g[t] + beta[t];
    float o1 = d1 * rstd * g[256 + t] + beta[256 + t];
    if (outF) { outF[base + t] = o0; outF[base + 256 + t] = o1; }
    if (outB) { outB[base + t] = (bf16)o0; outB[base + 256 + t] = (bf16)o1; }
}

// ================================================================ host orchestration
extern "C" void kernel_launch(void* const* d_in, const int* in_sizes, int n_in,
                              void* d_out, int out_size, void* d_ws, size_t ws_size,
                              hipStream_t stream) {
    const float* x        = (const float*)d_in[0];
    const float* txt      = (const float*)d_in[1];
    const unsigned char* vmask = (const unsigned char*)d_in[2];
    const unsigned char* tmask = (const unsigned char*)d_in[3];
    const float* wts_w    = (const float*)d_in[4];
    const float* wts_b    = (const float*)d_in[5];
    const float* ec_in_w  = (const float*)d_in[6];
    const float* ec_in_b  = (const float*)d_in[7];
    const float* ec_out_w = (const float*)d_in[8];
    const float* ec_out_b = (const float*)d_in[9];
    const float* cc_in_w  = (const float*)d_in[10];
    const float* cc_in_b  = (const float*)d_in[11];
    const float* cc_out_w = (const float*)d_in[12];
    const float* cc_out_b = (const float*)d_in[13];
    const float* ec_g = (const float*)d_in[14];
    const float* ec_bt = (const float*)d_in[15];
    const float* cc_g = (const float*)d_in[16];
    const float* cc_bt = (const float*)d_in[17];
    const float* fu_g = (const float*)d_in[18];
    const float* fu_bt = (const float*)d_in[19];
    const float* lin1_w = (const float*)d_in[20];
    const float* lin1_b = (const float*)d_in[21];
    const float* lin2_w = (const float*)d_in[22];
    const float* lin2_b = (const float*)d_in[23];
    (void)in_sizes; (void)n_in; (void)out_size; (void)ws_size;

    char* base = (char*)d_ws;
    size_t off = 0;
    auto alloc = [&](size_t bytes) -> void* {
        void* p = base + off;
        off += (bytes + 255) & ~(size_t)255;
        return p;
    };

    const int Lkmax = 1536;
    bf16* xb       = (bf16*)alloc((size_t)MTOK * DMODEL * 2);
    bf16* w_ec_in  = (bf16*)alloc((size_t)3 * DMODEL * DMODEL * 2);
    bf16* w_ec_out = (bf16*)alloc((size_t)DMODEL * DMODEL * 2);
    bf16* w_cc_in  = (bf16*)alloc((size_t)4 * 3 * DMODEL * DMODEL * 2);
    bf16* w_cc_out = (bf16*)alloc((size_t)4 * DMODEL * DMODEL * 2);
    bf16* w_lin1   = (bf16*)alloc((size_t)4 * DMODEL * DMODEL * 2);
    bf16* w_lin2   = (bf16*)alloc((size_t)4 * DMODEL * DMODEL * 2);
    float* sentf   = (float*)alloc((size_t)BATCH * DMODEL * 4);
    float* maskec  = (float*)alloc((size_t)BATCH * LLEN * 4);
    float* maskcc  = (float*)alloc((size_t)BATCH * Lkmax * 4);
    bf16* qkvb     = (bf16*)alloc((size_t)MTOK * 3 * DMODEL * 2);   // also cc KV (12288x1024)
    bf16* qprojb   = (bf16*)alloc((size_t)MTOK * DMODEL * 2);
    bf16* keysb    = (bf16*)alloc((size_t)Lkmax * BATCH * DMODEL * 2);
    bf16* qpk      = (bf16*)alloc((size_t)BATCH * NH * LLEN * HDIM * 2);
    bf16* kpk      = (bf16*)alloc((size_t)BATCH * NH * Lkmax * HDIM * 2);
    bf16* vpk      = (bf16*)alloc((size_t)BATCH * NH * Lkmax * HDIM * 2);
    float* obuf    = (float*)alloc((size_t)BATCH * NH * LLEN * HDIM * 4);
    bf16* packb    = (bf16*)alloc((size_t)MTOK * DMODEL * 2);
    float* ecproj  = (float*)alloc((size_t)MTOK * DMODEL * 4);
    float* totalf  = (float*)alloc((size_t)MTOK * DMODEL * 4);
    float* xecf    = (float*)alloc((size_t)MTOK * DMODEL * 4);
    float* xccf    = (float*)alloc((size_t)MTOK * DMODEL * 4);
    float* outf    = (float*)alloc((size_t)MTOK * DMODEL * 4);
    bf16* outb     = (bf16*)alloc((size_t)MTOK * DMODEL * 2);
    bf16* hb       = (bf16*)alloc((size_t)MTOK * 4 * DMODEL * 2);
    float* out2f   = (float*)alloc((size_t)MTOK * DMODEL * 4);

    auto cvt = [&](const float* src, bf16* dst, long n) {
        k_cvt<<<dim3((unsigned)((n + 255) / 256)), 256, 0, stream>>>(src, dst, n);
    };
    auto gemm = [&](const bf16* A, const bf16* W, const float* bias, const float* addC,
                    float* oF, bf16* oB, int M, int N, int K, int relu) {
        dim3 g((N + 127) / 128, (M + 63) / 64);
        k_gemm<<<g, 256, 0, stream>>>(A, W, bias, addC, oF, oB, M, N, K, relu);
    };

    // --- bf16 staging
    cvt(x, xb, (long)MTOK * DMODEL);
    cvt(ec_in_w, w_ec_in, (long)3 * DMODEL * DMODEL);
    cvt(ec_out_w, w_ec_out, (long)DMODEL * DMODEL);
    cvt(cc_in_w, w_cc_in, (long)4 * 3 * DMODEL * DMODEL);
    cvt(cc_out_w, w_cc_out, (long)4 * DMODEL * DMODEL);
    cvt(lin1_w, w_lin1, (long)4 * DMODEL * DMODEL);
    cvt(lin2_w, w_lin2, (long)4 * DMODEL * DMODEL);

    // --- text pooling + top-k relevance mask
    k_sent<<<BATCH, 256, 0, stream>>>(txt, wts_w, wts_b, tmask, sentf);
    k_topk<<<BATCH, 256, 0, stream>>>(x, sentf, vmask, maskec);

    const long nheadel = (long)BATCH * NH * LLEN * HDIM;
    const float scl = 0.125f; // 1/sqrt(64)

    // --- essential-context self attention (Lk = Lkpad = 1024)
    gemm(xb, w_ec_in, ec_in_b, nullptr, nullptr, qkvb, MTOK, 3 * DMODEL, DMODEL, 0);
    k_packq<<<dim3((unsigned)((nheadel + 255) / 256)), 256, 0, stream>>>(qkvb, 3 * DMODEL, 0, qpk, LLEN);
    k_packk<<<dim3((unsigned)((nheadel + 255) / 256)), 256, 0, stream>>>(qkvb, 3 * DMODEL, DMODEL, kpk, LLEN, LLEN);
    k_packvt<<<dim3((unsigned)((nheadel + 255) / 256)), 256, 0, stream>>>(qkvb, 3 * DMODEL, 2 * DMODEL, vpk, LLEN, LLEN);
    k_attn<<<dim3(LLEN / 16, BATCH * NH), 32, 0, stream>>>(qpk, kpk, vpk, maskec, obuf, LLEN, LLEN, LLEN, scl);
    k_packo<<<dim3((unsigned)(((long)MTOK * DMODEL + 255) / 256)), 256, 0, stream>>>(obuf, packb, LLEN);
    gemm(packb, w_ec_out, ec_out_b, nullptr, ecproj, nullptr, MTOK, DMODEL, DMODEL, 0);
    k_ln<<<MTOK, 256, 0, stream>>>(x, ecproj, ec_g, ec_bt, xecf, nullptr);

    // --- multi-stride pooled cross attentions
    const int s1g[4] = {1, 4, 16, 32}, s2g[4] = {2, 8, 24, 48};
    const int lp1g[4] = {1024, 256, 64, 32}, lp2g[4] = {512, 128, 43, 22};
    for (int g = 0; g < 4; ++g) {
        int Lk = lp1g[g] + lp2g[g];
        int Lkpad = (Lk + 63) & ~63;          // multiple of 64 for ping-pong attention loop
        int Mk = Lk * BATCH;
        long pooln = (long)Lk * BATCH * DMODEL;
        k_pool<<<dim3((unsigned)((pooln + 255) / 256)), 256, 0, stream>>>(
            x, vmask, keysb, maskcc, s1g[g], s2g[g], lp1g[g], Lk);
        const bf16* wg = w_cc_in + (size_t)g * 3 * DMODEL * DMODEL;
        gemm(xb, wg, cc_in_b + g * 3 * DMODEL, nullptr, nullptr, qprojb, MTOK, DMODEL, DMODEL, 0);
        gemm(keysb, wg + (size_t)DMODEL * DMODEL, cc_in_b + g * 3 * DMODEL + DMODEL,
             nullptr, nullptr, qkvb, Mk, 2 * DMODEL, DMODEL, 0);
        long nkv = (long)BATCH * NH * Lkpad * HDIM;
        k_packq<<<dim3((unsigned)((nheadel + 255) / 256)), 256, 0, stream>>>(qprojb, DMODEL, 0, qpk, LLEN);
        k_packk<<<dim3((unsigned)((nkv + 255) / 256)), 256, 0, stream>>>(qkvb, 2 * DMODEL, 0, kpk, Lk, Lkpad);
        k_packvt<<<dim3((unsigned)((nkv + 255) / 256)), 256, 0, stream>>>(qkvb, 2 * DMODEL, DMODEL, vpk, Lk, Lkpad);
        k_attn<<<dim3(LLEN / 16, BATCH * NH), 32, 0, stream>>>(qpk, kpk, vpk, maskcc, obuf, LLEN, Lk, Lkpad, scl);
        k_packo<<<dim3((unsigned)(((long)MTOK * DMODEL + 255) / 256)), 256, 0, stream>>>(obuf, packb, LLEN);
        gemm(packb, w_cc_out + (size_t)g * DMODEL * DMODEL, cc_out_b + g * DMODEL,
             (g == 0) ? nullptr : totalf, totalf, nullptr, MTOK, DMODEL, DMODEL, 0);
    }
    k_ln<<<MTOK, 256, 0, stream>>>(x, totalf, cc_g, cc_bt, xccf, nullptr);

    // --- fusion + FFN + final LN
    k_add<<<dim3((unsigned)(((long)MTOK * DMODEL + 255) / 256)), 256, 0, stream>>>(
        xecf, xccf, outf, outb, (long)MTOK * DMODEL);
    gemm(outb, w_lin1, lin1_b, nullptr, nullptr, hb, MTOK, 4 * DMODEL, DMODEL, 1);
    gemm(hb, w_lin2, lin2_b, nullptr, out2f, nullptr, MTOK, DMODEL, 4 * DMODEL, 0);
    k_ln<<<MTOK, 256, 0, stream>>>(outf, out2f, fu_g, fu_bt, (float*)d_out, nullptr);
}
